// GatedAttention_50861002719383
// MI455X (gfx1250) — compile-verified, matrix operand order checked by reading, not run on hardware
//
#include <hip/hip_runtime.h>
#include <hip/hip_bf16.h>

#define B_SZ   4
#define N_TOK  4096
#define DIM_   1024
#define HEADS_ 16
#define DHEAD  64
#define HID    1024
#define BH     (B_SZ * HEADS_)   /* 64 */
#define SCALE_ 0.125f            /* 64^-0.5 */

typedef __bf16 bf16;
typedef __attribute__((ext_vector_type(16))) __bf16 v16bf;
typedef __attribute__((ext_vector_type(8)))  __bf16 v8bf;
typedef __attribute__((ext_vector_type(8)))  float  v8f;

// ---------------- fragment loaders (wave32, 16x16x32 bf16 WMMA) ----------------
// A (16x32, MxK) per-lane layout (ISA 7.12.2): lane L holds row M=L&15;
// lanes 0-15 hold K=0..7 and 16..23, lanes 16-31 hold K=8..15 and 24..31.
// B (32x16, KxN) mirrored with N=L&15. Row-major source, contiguous K:
// two 16-byte loads per lane.
__device__ __forceinline__ v16bf frag_rm(const bf16* __restrict__ p, int ld, int lane) {
  const int r  = lane & 15;
  const int kb = (lane >> 4) << 3;          // 0 or 8
  const bf16* row = p + (size_t)r * ld;
  v8bf lo = *(const v8bf*)(row + kb);
  v8bf hi = *(const v8bf*)(row + 16 + kb);
  v16bf f;
#pragma unroll
  for (int i = 0; i < 8; ++i) { f[i] = lo[i]; f[i + 8] = hi[i]; }
  return f;
}

// Same pattern reading from LDS tiles staged as [16-rows][32 K contiguous].
__device__ __forceinline__ v16bf frag_lds(const bf16* p, int ld, int lane) {
  const int r  = lane & 15;
  const int kb = (lane >> 4) << 3;
  const bf16* row = p + r * ld;
  v8bf lo = *(const v8bf*)(row + kb);
  v8bf hi = *(const v8bf*)(row + 16 + kb);
  v16bf f;
#pragma unroll
  for (int i = 0; i < 8; ++i) { f[i] = lo[i]; f[i + 8] = hi[i]; }
  return f;
}

__device__ __forceinline__ v8f wmma_bf16(v16bf a, v16bf b, v8f c) {
  return __builtin_amdgcn_wmma_f32_16x16x32_bf16(false, a, false, b, (short)0, c, false, false);
}

__device__ __forceinline__ float elu1(float x) { return x > 0.f ? x + 1.f : __expf(x); }
__device__ __forceinline__ float sigm(float x) { return 1.f / (1.f + __expf(-x)); }

// ---------------- 0) f32 -> bf16 convert ----------------
__global__ void k_cvt_bf16(const float* __restrict__ s, bf16* __restrict__ d, int n) {
  int i = (blockIdx.x * blockDim.x + threadIdx.x) * 4;
  if (i + 3 < n) {
    float4 v = *(const float4*)(s + i);
    d[i]   = (bf16)v.x; d[i+1] = (bf16)v.y;
    d[i+2] = (bf16)v.z; d[i+3] = (bf16)v.w;
  }
}

// =======================================================================
// LDS ping-pong GEMM core: block = 128 thr (4 waves), tile M=128 N=64 K=32.
// Per k-step: fetch next A(128x32)+B(64x32) bf16 tiles into regs (6 b128
// loads/thread, in flight under the WMMAs), 8 WMMAs per wave from current
// LDS buffers, write staged regs to the other buffer, one barrier/step.
// =======================================================================
#define GEMM_LDSPIPE(APTR, BPTR, LDK)                                         \
  __shared__ __align__(16) bf16 Asm[2][128 * 32];                             \
  __shared__ __align__(16) bf16 Bsm[2][64 * 32];                              \
  v8bf ra[4], rb[2];                                                          \
  int rowA[4], qA[4], rowB[2], qB[2];                                         \
  _Pragma("unroll") for (int i = 0; i < 4; ++i) {                             \
    int c = tid + i * 128; rowA[i] = c >> 2; qA[i] = (c & 3) * 8;             \
  }                                                                           \
  _Pragma("unroll") for (int i = 0; i < 2; ++i) {                             \
    int c = tid + i * 128; rowB[i] = c >> 2; qB[i] = (c & 3) * 8;             \
  }                                                                           \
  _Pragma("unroll") for (int i = 0; i < 4; ++i)                               \
    ra[i] = *(const v8bf*)(APTR + (size_t)rowA[i] * (LDK) + qA[i]);           \
  _Pragma("unroll") for (int i = 0; i < 2; ++i)                               \
    rb[i] = *(const v8bf*)(BPTR + (size_t)rowB[i] * (LDK) + qB[i]);           \
  _Pragma("unroll") for (int i = 0; i < 4; ++i)                               \
    *(v8bf*)&Asm[0][rowA[i] * 32 + qA[i]] = ra[i];                            \
  _Pragma("unroll") for (int i = 0; i < 2; ++i)                               \
    *(v8bf*)&Bsm[0][rowB[i] * 32 + qB[i]] = rb[i];                            \
  __syncthreads();                                                            \
  int cur = 0;                                                                \
  v8f acc[2][4] = {};                                                         \
  for (int kb = 0; kb < (LDK); kb += 32) {                                    \
    const bool more = (kb + 32) < (LDK);                                      \
    if (more) {                                                               \
      _Pragma("unroll") for (int i = 0; i < 4; ++i)                           \
        ra[i] = *(const v8bf*)(APTR + (size_t)rowA[i] * (LDK) + kb + 32 + qA[i]); \
      _Pragma("unroll") for (int i = 0; i < 2; ++i)                           \
        rb[i] = *(const v8bf*)(BPTR + (size_t)rowB[i] * (LDK) + kb + 32 + qB[i]); \
    }                                                                         \
    v16bf af0 = frag_lds(&Asm[cur][(wave * 32) * 32], 32, lane);              \
    v16bf af1 = frag_lds(&Asm[cur][(wave * 32 + 16) * 32], 32, lane);         \
    _Pragma("unroll") for (int t = 0; t < 4; ++t) {                           \
      v16bf bfr = frag_lds(&Bsm[cur][(t * 16) * 32], 32, lane);               \
      acc[0][t] = wmma_bf16(af0, bfr, acc[0][t]);                             \
      acc[1][t] = wmma_bf16(af1, bfr, acc[1][t]);                             \
    }                                                                         \
    if (more) {                                                               \
      _Pragma("unroll") for (int i = 0; i < 4; ++i)                           \
        *(v8bf*)&Asm[cur ^ 1][rowA[i] * 32 + qA[i]] = ra[i];                  \
      _Pragma("unroll") for (int i = 0; i < 2; ++i)                           \
        *(v8bf*)&Bsm[cur ^ 1][rowB[i] * 32 + qB[i]] = rb[i];                  \
      __syncthreads();                                                        \
      cur ^= 1;                                                               \
    }                                                                         \
  }

// ---------------- 1) fused QKVG projection ----------------
// C[16384,4096] = Xb @ Wqkvg^T ; W rows are output features.
__global__ __launch_bounds__(128, 1)
void k_qkvg(const bf16* __restrict__ X, const bf16* __restrict__ W,
            const float* __restrict__ bg,
            bf16* __restrict__ Q, bf16* __restrict__ K,
            bf16* __restrict__ V, bf16* __restrict__ G) {
  const int tid = threadIdx.x, lane = tid & 31, wave = tid >> 5;
  const int mBlk = blockIdx.x * 128;
  const int o0   = blockIdx.y * 64;
  const bf16* aTile = X + (size_t)mBlk * DIM_;
  const bf16* bTile = W + (size_t)o0 * DIM_;
  GEMM_LDSPIPE(aTile, bTile, DIM_)

  const int m0  = mBlk + wave * 32;
  const int mo  = (lane >> 4) << 3;
  const int col = lane & 15;
#pragma unroll
  for (int t = 0; t < 4; ++t) {
    const int o   = o0 + t * 16 + col;
    const int cls = o >> 10;          // 0=q 1=k 2=v 3=g
    const int f   = o & 1023;
    const int h   = f >> 6, d = f & 63;
#pragma unroll
    for (int rt = 0; rt < 2; ++rt)
#pragma unroll
      for (int i = 0; i < 8; ++i) {
        const int m = m0 + rt * 16 + i + mo;
        const int bb = m >> 12, n = m & (N_TOK - 1);
        const size_t idx = ((size_t)(bb * HEADS_ + h) * N_TOK + n) * DHEAD + d;
        float val = acc[rt][t][i];
        if (cls == 0)      Q[idx] = (bf16)elu1(val);
        else if (cls == 1) K[idx] = (bf16)elu1(val);
        else if (cls == 2) V[idx] = (bf16)val;
        else               G[idx] = (bf16)sigm(val + bg[f]);
      }
  }
}

// ---------------- 2) kv = K^T V per (b,h), stored transposed; + k_sum ---------
// 32-token tiles staged in LDS transposed ([d][n], [e][n]) via coalesced
// global_load_b128 + ds_store_b16 scatter; fragments are ds_load_b128 pairs.
__global__ void k_kv(const bf16* __restrict__ K, const bf16* __restrict__ V,
                     bf16* __restrict__ KVT, float* __restrict__ KS) {
  const int bh   = blockIdx.x;
  const int tid  = threadIdx.x;
  const int lane = tid & 31;
  const int wave = tid >> 5;
  const bf16* kp = K + (size_t)bh * N_TOK * DHEAD;
  const bf16* vp = V + (size_t)bh * N_TOK * DHEAD;

  { // k_sum[d]
    const int d = tid & 63, half = tid >> 6;
    float s = 0.f;
    for (int n = half; n < N_TOK; n += 2) s += (float)kp[(size_t)n * DHEAD + d];
    __shared__ float sm[128];
    sm[tid] = s;
    __syncthreads();
    if (tid < 64) KS[bh * DHEAD + d] = sm[tid] + sm[tid + 64];
    __syncthreads();
  }

  __shared__ __align__(16) bf16 Klds[DHEAD * 32];   // [d][n]
  __shared__ __align__(16) bf16 Vlds[DHEAD * 32];   // [e][n]

  const int dtile = wave * 16;
  v8f acc[4] = {};
  for (int n0 = 0; n0 < N_TOK; n0 += 32) {
#pragma unroll
    for (int cc = 0; cc < 2; ++cc) {
      const int chunk = tid + cc * 128;        // 0..255
      const int n = chunk >> 3;
      const int c = chunk & 7;
      v8bf kk = *(const v8bf*)(kp + (size_t)(n0 + n) * DHEAD + c * 8);
      v8bf vv = *(const v8bf*)(vp + (size_t)(n0 + n) * DHEAD + c * 8);
#pragma unroll
      for (int j = 0; j < 8; ++j) {
        Klds[(c * 8 + j) * 32 + n] = kk[j];
        Vlds[(c * 8 + j) * 32 + n] = vv[j];
      }
    }
    __syncthreads();
    v16bf a = frag_lds(Klds + dtile * 32, 32, lane);
#pragma unroll
    for (int t = 0; t < 4; ++t) {
      v16bf b = frag_lds(Vlds + t * 16 * 32, 32, lane);
      acc[t] = wmma_bf16(a, b, acc[t]);
    }
    __syncthreads();
  }
  const int mo = (lane >> 4) << 3, col = lane & 15;
#pragma unroll
  for (int t = 0; t < 4; ++t)
#pragma unroll
    for (int i = 0; i < 8; ++i) {
      const int d = dtile + i + mo;
      const int e = t * 16 + col;
      KVT[(size_t)bh * DHEAD * DHEAD + (size_t)e * DHEAD + d] = (bf16)acc[t][i];
    }
}

// ---------------- 3) z = 1 / max(q . k_sum, 1e-6) ----------------
__global__ void k_z(const bf16* __restrict__ Q, const float* __restrict__ KS,
                    float* __restrict__ Z) {
  const int idx = blockIdx.x * blockDim.x + threadIdx.x;   // bh*N + n
  const int bh  = idx >> 12;
  const bf16*  qp = Q + (size_t)idx * DHEAD;
  const float* ks = KS + bh * DHEAD;
  float dot = 0.f;
#pragma unroll 8
  for (int d = 0; d < DHEAD; ++d) dot += (float)qp[d] * ks[d];
  Z[idx] = 1.0f / fmaxf(dot, 1e-6f);
}

// ---------------- 4) out = (q*scale) @ kv * z * g ----------------
// kv stored transposed [e][d] -> B fragments are plain row-major loads.
__global__ void k_attn(const bf16* __restrict__ Q, const bf16* __restrict__ KVT,
                       const float* __restrict__ Z, const bf16* __restrict__ G,
                       bf16* __restrict__ A) {
  const int bh   = blockIdx.y;
  const int lane = threadIdx.x & 31;
  const int n0   = blockIdx.x * 64 + (threadIdx.x >> 5) * 16;
  const bf16* qp = Q + ((size_t)bh * N_TOK + n0) * DHEAD;
  const bf16* kv = KVT + (size_t)bh * DHEAD * DHEAD;
  v8f acc[4] = {};
#pragma unroll
  for (int kb = 0; kb < DHEAD; kb += 32) {
    v16bf a = frag_rm(qp + kb, DHEAD, lane);
#pragma unroll
    for (int t = 0; t < 4; ++t) {
      v16bf b = frag_rm(kv + (size_t)(t * 16) * DHEAD + kb, DHEAD, lane);
      acc[t] = wmma_bf16(a, b, acc[t]);
    }
  }
  const int bb = bh >> 4, h = bh & 15;
  const int mo = (lane >> 4) << 3, col = lane & 15;
#pragma unroll
  for (int t = 0; t < 4; ++t)
#pragma unroll
    for (int i = 0; i < 8; ++i) {
      const int n = n0 + i + mo;
      float val = acc[t][i] * SCALE_ * Z[bh * N_TOK + n];
      val *= (float)G[((size_t)bh * N_TOK + n) * DHEAD + t * 16 + col];
      A[((size_t)(bb * N_TOK + n)) * HID + h * DHEAD + t * 16 + col] = (bf16)val;
    }
}

// ---------------- 5) final projection: O = A @ Wo^T ----------------
__global__ __launch_bounds__(128, 1)
void k_out(const bf16* __restrict__ Ain, const bf16* __restrict__ Wo,
           float* __restrict__ O) {
  const int tid = threadIdx.x, lane = tid & 31, wave = tid >> 5;
  const int mBlk = blockIdx.x * 128;
  const int o0   = blockIdx.y * 64;
  const bf16* aTile = Ain + (size_t)mBlk * HID;
  const bf16* bTile = Wo + (size_t)o0 * HID;
  GEMM_LDSPIPE(aTile, bTile, HID)

  const int m0 = mBlk + wave * 32;
  const int mo = (lane >> 4) << 3, col = lane & 15;
#pragma unroll
  for (int rt = 0; rt < 2; ++rt)
#pragma unroll
    for (int t = 0; t < 4; ++t)
#pragma unroll
      for (int i = 0; i < 8; ++i)
        O[(size_t)(m0 + rt * 16 + i + mo) * DIM_ + o0 + t * 16 + col] = acc[rt][t][i];
}

// ---------------- host ----------------
extern "C" void kernel_launch(void* const* d_in, const int* in_sizes, int n_in,
                              void* d_out, int out_size, void* d_ws, size_t ws_size,
                              hipStream_t stream) {
  const float* x  = (const float*)d_in[0];
  const float* Wq = (const float*)d_in[1];
  const float* Wk = (const float*)d_in[2];
  const float* Wv = (const float*)d_in[3];
  const float* Wg = (const float*)d_in[4];
  const float* bg = (const float*)d_in[5];
  const float* Wo = (const float*)d_in[6];
  float* out = (float*)d_out;

  char* p = (char*)d_ws;
  bf16* xb   = (bf16*)p;  p += (size_t)B_SZ * N_TOK * DIM_ * 2;
  bf16* wb   = (bf16*)p;  p += (size_t)4 * HID * DIM_ * 2;     // Wq|Wk|Wv|Wg
  bf16* wob  = (bf16*)p;  p += (size_t)DIM_ * HID * 2;
  bf16* qbuf = (bf16*)p;  p += (size_t)BH * N_TOK * DHEAD * 2;
  bf16* kbuf = (bf16*)p;  p += (size_t)BH * N_TOK * DHEAD * 2;
  bf16* vbuf = (bf16*)p;  p += (size_t)BH * N_TOK * DHEAD * 2;
  bf16* gbuf = (bf16*)p;  p += (size_t)BH * N_TOK * DHEAD * 2;
  bf16* kvb  = (bf16*)p;  p += (size_t)BH * DHEAD * DHEAD * 2;
  float* ks  = (float*)p; p += (size_t)BH * DHEAD * 4;
  float* zb  = (float*)p; p += (size_t)BH * N_TOK * 4;
  bf16* ab   = (bf16*)p;  p += (size_t)B_SZ * N_TOK * HID * 2;

  const int nX = B_SZ * N_TOK * DIM_;     // 16,777,216
  const int nW = HID * DIM_;              //  1,048,576
  k_cvt_bf16<<<nX / 4 / 256, 256, 0, stream>>>(x,  xb,          nX);
  k_cvt_bf16<<<nW / 4 / 256, 256, 0, stream>>>(Wq, wb + 0 * (size_t)nW, nW);
  k_cvt_bf16<<<nW / 4 / 256, 256, 0, stream>>>(Wk, wb + 1 * (size_t)nW, nW);
  k_cvt_bf16<<<nW / 4 / 256, 256, 0, stream>>>(Wv, wb + 2 * (size_t)nW, nW);
  k_cvt_bf16<<<nW / 4 / 256, 256, 0, stream>>>(Wg, wb + 3 * (size_t)nW, nW);
  k_cvt_bf16<<<nW / 4 / 256, 256, 0, stream>>>(Wo, wob,         nW);

  k_qkvg<<<dim3(B_SZ * N_TOK / 128, 4 * HID / 64), 128, 0, stream>>>(
      xb, wb, bg, qbuf, kbuf, vbuf, gbuf);
  k_kv<<<BH, 128, 0, stream>>>(kbuf, vbuf, kvb, ks);
  k_z<<<BH * N_TOK / 256, 256, 0, stream>>>(qbuf, ks, zb);
  k_attn<<<dim3(N_TOK / 64, BH), 128, 0, stream>>>(qbuf, kvb, zb, gbuf, ab);
  k_out<<<dim3(B_SZ * N_TOK / 128, DIM_ / 64), 128, 0, stream>>>(ab, wob, out);
}
